// TopDownAttention_48979807044177
// MI455X (gfx1250) — matrix operand order repaired and checked by reading, not compiled, not run on hardware
//
#include <hip/hip_runtime.h>
#include <hip/hip_bf16.h>
#include <math.h>

// Problem constants (from reference): B=64, O=100, F1=2048, F2=1024, A=512
// M = B*O = 6400 rows, K = F1+F2 = 3072, N = A = 512.

typedef __attribute__((ext_vector_type(16))) __bf16 v16bf;
typedef __attribute__((ext_vector_type(8)))  float  v8f;

union FragBits {
    uint4 u[2];
    v16bf v;
};

__device__ inline unsigned short f2bf(float f) {
    // round-to-nearest-even f32 -> bf16
    unsigned u = __float_as_uint(f);
    u += 0x7FFFu + ((u >> 16) & 1u);
    return (unsigned short)(u >> 16);
}

__device__ inline v8f wmma_bf16(v16bf a, v16bf b, v8f c) {
    // D = A(16x32 bf16) * B(32x16 bf16) + C(16x16 f32)
    return __builtin_amdgcn_wmma_f32_16x16x32_bf16(
        /*neg_a=*/false, a, /*neg_b=*/false, b,
        /*c_mod=*/(short)0, c, /*reuse_a=*/false, /*reuse_b=*/false);
}

// ---------------------------------------------------------------------------
// Kernel 0: repack w_tanh / w_gate (f32, 3072x512 row-major, K-major) into
// per-lane WMMA B fragments in bf16.
// Fragment (kt, nt): 32 K-rows x 16 N-cols. Lane l holds column n = nt*16 + (l&15).
//   lanes 0-15 : dword i = {K=2i, K=2i+1}        (K 0..15 of the tile)
//   lanes 16-31: dword i = {K=16+2i, K=17+2i}    (K 16..31 of the tile)
// Storage: wpk[ ((sel*96 + kt)*32 + nt)*256 + lane*8 + i ]  (dwords)
// ---------------------------------------------------------------------------
__global__ __launch_bounds__(256) void repack_weights(
    const float* __restrict__ w_tanh,
    const float* __restrict__ w_gate,
    unsigned*    __restrict__ wpk)
{
    int id   = blockIdx.x * 256 + threadIdx.x;  // 0 .. 196607
    int lane = id & 31;
    int frag = id >> 5;                         // 0 .. 6143
    int nt   = frag & 31;
    int kts  = frag >> 5;                       // 0 .. 191 (sel*96 + kt)
    int kt   = (kts < 96) ? kts : (kts - 96);
    const float* __restrict__ src = (kts < 96) ? w_tanh : w_gate;

    int n     = nt * 16 + (lane & 15);
    int kbase = kt * 32 + ((lane >> 4) << 4);   // +16 for upper half-wave
    unsigned* dst = wpk + (size_t)frag * 256 + lane * 8;

#pragma unroll
    for (int i = 0; i < 8; ++i) {
        int k = kbase + 2 * i;
        float x0 = src[(size_t)k * 512 + n];
        float x1 = src[(size_t)(k + 1) * 512 + n];
        dst[i] = (unsigned)f2bf(x0) | ((unsigned)f2bf(x1) << 16);
    }
}

// ---------------------------------------------------------------------------
// Kernel 1: fused (x @ w_tanh), (x @ w_gate), gated-tanh, dot w_score -> scores
// One block = 16 rows of x. x row m0+m = concat(img[b,o,:], ques[b,:]),
// b = M/100, o = M%100. Full K (3072) staged in LDS as packed bf16 (96 KB).
// 8 waves; wave w owns N columns [w*64, w*64+64) for BOTH weight matrices.
// ---------------------------------------------------------------------------
__global__ __launch_bounds__(256) void fused_gemm_score(
    const float*    __restrict__ img,     // (64,100,2048)
    const float*    __restrict__ ques,    // (64,1024)
    const unsigned* __restrict__ wpk,     // packed bf16 weight fragments
    const float*    __restrict__ b_tanh,  // (512)
    const float*    __restrict__ b_gate,  // (512)
    const float*    __restrict__ w_score, // (512)
    const float*    __restrict__ b_score, // (1)
    float*          __restrict__ scores)  // (6400)
{
    __shared__ unsigned ldsA[16 * 1536];  // 16 rows x 3072 bf16, packed pairs
    __shared__ float    ldsS[16];

    const int tid = threadIdx.x;
    const int m0  = blockIdx.x * 16;

    // ---- Stage A tile: 16 rows x 3072 f32 -> packed bf16 pairs in LDS ----
    for (int idx = tid; idx < 16 * 1536; idx += 256) {
        int m  = idx / 1536;
        int kd = idx - m * 1536;
        int M  = m0 + m;
        int b  = M / 100;
        int o  = M - b * 100;
        int k  = kd * 2;
        float x0, x1;
        if (k < 2048) {
            const float* r = img + ((size_t)(b * 100 + o)) * 2048;
            x0 = r[k];
            x1 = r[k + 1];
        } else {
            const float* r = ques + (size_t)b * 1024 + (k - 2048);
            x0 = r[0];
            x1 = r[1];
        }
        ldsA[idx] = (unsigned)f2bf(x0) | ((unsigned)f2bf(x1) << 16);
    }
    if (tid < 16) ldsS[tid] = b_score[0];
    __syncthreads();

    const int lane = tid & 31;
    const int w    = tid >> 5;  // wave id 0..7

    v8f acc_t[4], acc_g[4];
#pragma unroll
    for (int j = 0; j < 4; ++j) { acc_t[j] = {}; acc_g[j] = {}; }

    // A-fragment LDS base for this lane (contiguous k order, same as the B packing):
    //   row m = lane&15 ; lanes>=16 hold K 16..31 of the 32-K step.
    const int a_row  = lane & 15;
    const int a_koff = (lane >> 4) << 3;  // dword offset: 0 or 8

    for (int ks = 0; ks < 96; ++ks) {
        FragBits fa;
        const unsigned* ap = ldsA + a_row * 1536 + ks * 16 + a_koff;
        fa.u[0] = *(const uint4*)(ap);
        fa.u[1] = *(const uint4*)(ap + 4);
        v16bf a = fa.v;

#pragma unroll
        for (int j = 0; j < 4; ++j) {
            int nt = w * 4 + j;
            // w_tanh fragment (sel=0)
            {
                const unsigned* bp =
                    wpk + ((size_t)(ks * 32 + nt)) * 256 + lane * 8;
                FragBits fb;
                fb.u[0] = *(const uint4*)(bp);
                fb.u[1] = *(const uint4*)(bp + 4);
                acc_t[j] = wmma_bf16(a, fb.v, acc_t[j]);
            }
            // w_gate fragment (sel=1)
            {
                const unsigned* bp =
                    wpk + ((size_t)((96 + ks) * 32 + nt)) * 256 + lane * 8;
                FragBits fb;
                fb.u[0] = *(const uint4*)(bp);
                fb.u[1] = *(const uint4*)(bp + 4);
                acc_g[j] = wmma_bf16(a, fb.v, acc_g[j]);
            }
        }
    }

    // ---- Epilogue: gated tanh, weight by w_score, reduce over N ----
    // C/D layout: lanes 0-15 -> N=lane, M=vgpr; lanes 16-31 -> N=lane-16, M=vgpr+8
    float partial[8];
#pragma unroll
    for (int v = 0; v < 8; ++v) partial[v] = 0.0f;

    const int nl = lane & 15;
#pragma unroll
    for (int j = 0; j < 4; ++j) {
        int n = (w * 4 + j) * 16 + nl;
        float bt = b_tanh[n];
        float bg = b_gate[n];
        float ws = w_score[n];
#pragma unroll
        for (int v = 0; v < 8; ++v) {
            float t = tanhf(acc_t[j][v] + bt);
            float s = 1.0f / (1.0f + __expf(-(acc_g[j][v] + bg)));
            partial[v] += t * s * ws;
        }
    }

    const int mofs = (lane >> 4) << 3;  // 0 or 8
#pragma unroll
    for (int v = 0; v < 8; ++v)
        atomicAdd(&ldsS[v + mofs], partial[v]);  // ds_add_f32
    __syncthreads();

    if (tid < 16) scores[m0 + tid] = ldsS[tid];
}

// ---------------------------------------------------------------------------
// Kernel 2: per-batch masked softmax over O=100 + attn-weighted pooling.
// ---------------------------------------------------------------------------
__global__ __launch_bounds__(256) void softmax_pool(
    const float* __restrict__ scores,   // (6400)
    const int*   __restrict__ num_obj,  // (64)
    const float* __restrict__ img,      // (64,100,2048)
    float*       __restrict__ out)      // (64,2048)
{
    __shared__ float sh[100];
    __shared__ float red;

    const int b   = blockIdx.x;
    const int tid = threadIdx.x;
    const int no  = num_obj[b];

    if (tid < 100) {
        float s = scores[b * 100 + tid];
        sh[tid] = (tid >= no) ? -INFINITY : s;
    }
    __syncthreads();

    if (tid == 0) {
        float mx = -INFINITY;
        for (int o = 0; o < 100; ++o) mx = fmaxf(mx, sh[o]);
        red = mx;
    }
    __syncthreads();
    float mx = red;
    if (tid < 100) sh[tid] = __expf(sh[tid] - mx);
    __syncthreads();
    if (tid == 0) {
        float sum = 0.0f;
        for (int o = 0; o < 100; ++o) sum += sh[o];
        red = 1.0f / sum;
    }
    __syncthreads();
    float inv = red;
    if (tid < 100) sh[tid] *= inv;
    __syncthreads();

    const float* base = img + (size_t)b * 100 * 2048;
    for (int f = tid; f < 2048; f += 256) {
        float acc = 0.0f;
        for (int o = 0; o < 100; ++o)
            acc += sh[o] * base[(size_t)o * 2048 + f];
        out[(size_t)b * 2048 + f] = acc;
    }
}

// ---------------------------------------------------------------------------
extern "C" void kernel_launch(void* const* d_in, const int* in_sizes, int n_in,
                              void* d_out, int out_size, void* d_ws, size_t ws_size,
                              hipStream_t stream)
{
    const float* img     = (const float*)d_in[0];  // (64,100,2048)
    const float* ques    = (const float*)d_in[1];  // (64,1024)
    const int*   num_obj = (const int*)  d_in[2];  // (64)
    const float* w_tanh  = (const float*)d_in[3];  // (3072,512)
    const float* b_tanh  = (const float*)d_in[4];  // (512)
    const float* w_gate  = (const float*)d_in[5];  // (3072,512)
    const float* b_gate  = (const float*)d_in[6];  // (512)
    const float* w_score = (const float*)d_in[7];  // (512)
    const float* b_score = (const float*)d_in[8];  // (1)
    float* out = (float*)d_out;                    // (64,2048)

    // Workspace layout: [packed bf16 weights: 6,291,456 B][scores: 6400 f32]
    unsigned* wpk    = (unsigned*)d_ws;
    float*    scores = (float*)((char*)d_ws + (size_t)2 * 96 * 32 * 256 * 4);

    // 2 matrices * 96 kt * 32 nt * 32 lanes = 196608 lane-fragments
    repack_weights<<<768, 256, 0, stream>>>(w_tanh, w_gate, wpk);
    fused_gemm_score<<<400, 256, 0, stream>>>(img, ques, wpk, b_tanh, b_gate,
                                              w_score, b_score, scores);
    softmax_pool<<<64, 256, 0, stream>>>(scores, num_obj, img, out);
}
